// SinkhornAttention_26465588478197
// MI455X (gfx1250) — matrix operand order repaired and checked by reading, not compiled, not run on hardware
//
#include <hip/hip_runtime.h>
#include <hip/hip_bf16.h>

// ---- problem constants (b=2,h=16,t=8192,d=64, bucket=256) ----
#define TT   8192
#define DD   64
#define BS   256
#define NB   32      // buckets = kv_buckets = T/BS
#define BHN  32      // b*h

typedef __attribute__((ext_vector_type(16))) _Float16 v16h;
typedef __attribute__((ext_vector_type(8)))  _Float16 v8h;
typedef __attribute__((ext_vector_type(8)))  float    v8f;

// ---- LDS layout for attention kernel (units: halves) ----
#define QPAD 72                     // row stride for sQ / sK (64 + 8 pad)
#define VPAD 520                    // row stride for sVt (512 + 8 pad)
#define PPAD 40                     // row stride for per-wave P scratch (32 + 8)
#define SM_Q 0
#define SM_K (SM_Q + 256 * QPAD)
#define SM_V (SM_K + 512 * QPAD)    // sVt: 80 rows (64 data + 16 "ones" rows)
#define SM_P (SM_V + 80  * VPAD)
#define SMEM_HALVES (SM_P + 8 * 32 * PPAD)
#define SMEM_BYTES  (SMEM_HALVES * 2)

// A-fragment of a 16x32 f16 tile from a row-major LDS row:
// element j lives at d = (j&7) + 16*(j>>3) + 8*half  (ISA 7.12.2)
__device__ __forceinline__ v16h frag_ld_A(const _Float16* p, int hf) {
  v8h lo = *(const v8h*)(p + 8 * hf);
  v8h hi = *(const v8h*)(p + 16 + 8 * hf);
  v16h r;
#pragma unroll
  for (int i = 0; i < 8; ++i) { r[i] = lo[i]; r[i + 8] = hi[i]; }
  return r;
}

// B-fragment of a 32x16 f16 tile: element j lives at k = (j&7) + 16*(j>>3) + 8*half
__device__ __forceinline__ v16h frag_ld_B(const _Float16* p, int hf) {
  v8h lo = *(const v8h*)(p + 8 * hf);
  v8h hi = *(const v8h*)(p + 16 + 8 * hf);
  v16h r;
#pragma unroll
  for (int i = 0; i < 8; ++i) { r[i] = lo[i]; r[i + 8] = hi[i]; }
  return r;
}

// ---- VALU cross-lane max-reduce over 16-lane groups (DPP, no LDS traffic) ----
#define DPP8_SEL(a,b,c,d,e,f,g,h) \
  ((a)|((b)<<3)|((c)<<6)|((d)<<9)|((e)<<12)|((f)<<15)|((g)<<18)|((h)<<21))

__device__ __forceinline__ float max_x16(float x) {
  int v;
  // xor 1 within 8 lanes
  v = __builtin_amdgcn_mov_dpp8(__float_as_int(x), DPP8_SEL(1,0,3,2,5,4,7,6));
  x = fmaxf(x, __int_as_float(v));
  // xor 2
  v = __builtin_amdgcn_mov_dpp8(__float_as_int(x), DPP8_SEL(2,3,0,1,6,7,4,5));
  x = fmaxf(x, __int_as_float(v));
  // xor 4
  v = __builtin_amdgcn_mov_dpp8(__float_as_int(x), DPP8_SEL(4,5,6,7,0,1,2,3));
  x = fmaxf(x, __int_as_float(v));
  // cross the 8-lane halves of the 16-row: ROW_MIRROR (0x140) reverses 16 lanes;
  // since each 8-group is already uniform, this is an xor-8 exchange.
  v = __builtin_amdgcn_update_dpp(0, __float_as_int(x), 0x140, 0xf, 0xf, true);
  return fmaxf(x, __int_as_float(v));
}

// ---------------- Phase 1: sort-net routing (topk=1) ----------------
__global__ void sinkhorn_route_kernel(const float* __restrict__ q,
                                      const float* __restrict__ k,
                                      int* __restrict__ ridx,
                                      float* __restrict__ rval) {
  __shared__ float smQ[NB][DD];
  __shared__ float smK[NB][DD];
  __shared__ float sc[NB][NB];
  const int bh = blockIdx.x;
  const int t  = threadIdx.x;           // 256 threads
  const float* qb = q + (size_t)bh * TT * DD;
  const float* kb = k + (size_t)bh * TT * DD;

  for (int p = t; p < NB * DD; p += 256) {
    const int u = p >> 6, d = p & 63;
    const float* s0 = qb + (size_t)u * BS * DD + d;
    const float* s1 = kb + (size_t)u * BS * DD + d;
    float a = 0.f, b = 0.f;
    for (int tok = 0; tok < BS; ++tok) { a += s0[tok * DD]; b += s1[tok * DD]; }
    smQ[u][d] = a * (1.0f / BS);
    smK[u][d] = b * (1.0f / BS);
  }
  __syncthreads();

  for (int p = t; p < NB * NB; p += 256) {
    const int i = p >> 5, j = p & 31;
    float s = 0.f;
    for (int d = 0; d < DD; ++d) s += smQ[i][d] * smK[j][d];
    sc[i][j] = s * 0.125f;
  }
  __syncthreads();

  if (t < NB) {
    float mx = -3.0e38f; int am = 0;
    for (int j = 0; j < NB; ++j) {
      const float s = sc[t][j];
      if (s > mx) { mx = s; am = j; }
    }
    float se = 0.f;
    for (int j = 0; j < NB; ++j) se += __expf(sc[t][j] - mx);
    ridx[bh * NB + t] = am;
    rval[bh * NB + t] = 1.0f / se;
  }
}

// ---------------- Phase 2: bucketed flash attention via WMMA ----------------
__global__ __launch_bounds__(256, 1)
void sinkhorn_attn_kernel(const float* __restrict__ q,
                          const float* __restrict__ k,
                          const float* __restrict__ v,
                          const int* __restrict__ ridx,
                          const float* __restrict__ rval,
                          float* __restrict__ out) {
  extern __shared__ _Float16 sm[];
  _Float16* sQ  = sm + SM_Q;   // [256][QPAD] q * 0.125, f16
  _Float16* sK  = sm + SM_K;   // [512][QPAD] rows 0-255: val*K[route]; 256-511: K[local]
  _Float16* sVt = sm + SM_V;   // [80][VPAD]  transposed V_cat; rows 64-79 = 1.0 (row-sum tile)
  _Float16* sP  = sm + SM_P;   // [8 waves][32][PPAD] per-wave P transpose scratch

  const int u   = blockIdx.x;
  const int bh  = blockIdx.y;
  const int tid = threadIdx.x;

  const int   route = ridx[bh * NB + u];
  const float rv    = rval[bh * NB + u];

  const float* qb = q + ((size_t)bh * NB + u) * BS * DD;
  const float* kl = k + ((size_t)bh * NB + u) * BS * DD;
  const float* kr = k + ((size_t)bh * NB + route) * BS * DD;
  const float* vl = v + ((size_t)bh * NB + u) * BS * DD;

  // ---- ones rows for the row-sum accumulator tile (d = 64..79) ----
  for (int i = tid; i < 16 * VPAD; i += 256) sVt[64 * VPAD + i] = (_Float16)1.0f;

  // ---- stage tiles into LDS as f16 ----
  for (int i = tid * 4; i < BS * DD; i += 256 * 4) {
    const int row = i >> 6, d = i & 63;
    // Q, pre-scaled by softmax scale
    float4 f = *(const float4*)(qb + i);
    _Float16* dq = sQ + row * QPAD + d;
    dq[0] = (_Float16)(f.x * 0.125f); dq[1] = (_Float16)(f.y * 0.125f);
    dq[2] = (_Float16)(f.z * 0.125f); dq[3] = (_Float16)(f.w * 0.125f);
    // K_cat row-major: routed (scaled by val) then local
    f = *(const float4*)(kr + i);
    _Float16* dk = sK + row * QPAD + d;
    dk[0] = (_Float16)(f.x * rv); dk[1] = (_Float16)(f.y * rv);
    dk[2] = (_Float16)(f.z * rv); dk[3] = (_Float16)(f.w * rv);
    // V_cat transposed, col 0-255 = val*K[route] (reference's b_v_r = b_k_r)
    sVt[(d + 0) * VPAD + row] = dk[0];
    sVt[(d + 1) * VPAD + row] = dk[1];
    sVt[(d + 2) * VPAD + row] = dk[2];
    sVt[(d + 3) * VPAD + row] = dk[3];
    f = *(const float4*)(kl + i);
    dk = sK + (row + 256) * QPAD + d;
    dk[0] = (_Float16)f.x; dk[1] = (_Float16)f.y;
    dk[2] = (_Float16)f.z; dk[3] = (_Float16)f.w;
    f = *(const float4*)(vl + i);
    sVt[(d + 0) * VPAD + 256 + row] = (_Float16)f.x;
    sVt[(d + 1) * VPAD + 256 + row] = (_Float16)f.y;
    sVt[(d + 2) * VPAD + 256 + row] = (_Float16)f.z;
    sVt[(d + 3) * VPAD + 256 + row] = (_Float16)f.w;
  }
  __syncthreads();

  // ---- per-wave flash attention over 512 keys ----
  const int wv    = tid >> 5;
  const int lane  = tid & 31;
  const int hf    = lane >> 4;   // half-wave select
  const int n     = lane & 15;   // column / row-in-tile index
  const int rows0 = wv * 32;     // this wave's Q rows

  // Q A-fragments: 2 M-tiles x 2 K-steps (K = 64 -> 2 x 32)
  v16h aQ[2][2];
#pragma unroll
  for (int rt = 0; rt < 2; ++rt)
#pragma unroll
    for (int ks = 0; ks < 2; ++ks)
      aQ[rt][ks] = frag_ld_A(sQ + (rows0 + rt * 16 + n) * QPAD + ks * 32, hf);

  // O tiles: ct 0..3 = output d-columns, ct 4 = replicated row-sum (softmax denom)
  v8f   O[2][5];
  float mi[2][8];
#pragma unroll
  for (int rt = 0; rt < 2; ++rt) {
#pragma unroll
    for (int ct = 0; ct < 5; ++ct) O[rt][ct] = v8f{};
#pragma unroll
    for (int r = 0; r < 8; ++r) mi[rt][r] = -3.0e38f;
  }

  _Float16* sPw = sP + wv * 32 * PPAD;

  for (int jc = 0; jc < 16; ++jc) {   // 32 keys per chunk
    const int j0 = jc * 32;

    // K B-fragments (shared across both M-tiles)
    v16h bK[2][2];
#pragma unroll
    for (int nt = 0; nt < 2; ++nt)
#pragma unroll
      for (int ks = 0; ks < 2; ++ks)
        bK[nt][ks] = frag_ld_B(sK + (j0 + nt * 16 + n) * QPAD + ks * 32, hf);

    // S = Q * K_cat^T (pre-scaled)
    v8f S[2][2];
#pragma unroll
    for (int rt = 0; rt < 2; ++rt)
#pragma unroll
      for (int nt = 0; nt < 2; ++nt) {
        v8f c = {};
        c = __builtin_amdgcn_wmma_f32_16x16x32_f16(false, aQ[rt][0], false, bK[nt][0], (short)0, c, false, false);
        c = __builtin_amdgcn_wmma_f32_16x16x32_f16(false, aQ[rt][1], false, bK[nt][1], (short)0, c, false, false);
        S[rt][nt] = c;
      }

    // online softmax: row max via DPP (pure VALU); row sums come from the
    // ones-column WMMA tile, so no sum reduction is needed here at all.
#pragma unroll
    for (int rt = 0; rt < 2; ++rt) {
#pragma unroll
      for (int r = 0; r < 8; ++r) {
        const float tmx   = max_x16(fmaxf(S[rt][0][r], S[rt][1][r]));
        const float nm    = fmaxf(mi[rt][r], tmx);
        const float alpha = __expf(mi[rt][r] - nm);
        mi[rt][r] = nm;
        const float p0 = __expf(S[rt][0][r] - nm);
        const float p1 = __expf(S[rt][1][r] - nm);
#pragma unroll
        for (int ct = 0; ct < 5; ++ct) O[rt][ct][r] *= alpha;
        // stash P (C-layout -> row-major scratch) for A-fragment reload
        const int prow = rt * 16 + r + 8 * hf;
        sPw[prow * PPAD + n]      = (_Float16)p0;
        sPw[prow * PPAD + 16 + n] = (_Float16)p1;
      }
    }

    // O += P * V_cat ; ct=4 multiplies the all-ones rows -> accumulates the
    // softmax denominator with the same alpha-rescaled recurrence as O.
    v16h aP[2];
#pragma unroll
    for (int rt = 0; rt < 2; ++rt)
      aP[rt] = frag_ld_A(sPw + (rt * 16 + n) * PPAD, hf);
#pragma unroll
    for (int ct = 0; ct < 5; ++ct) {
      const v16h bV = frag_ld_B(sVt + (ct * 16 + n) * VPAD + j0, hf);
#pragma unroll
      for (int rt = 0; rt < 2; ++rt)
        O[rt][ct] = __builtin_amdgcn_wmma_f32_16x16x32_f16(false, aP[rt], false, bV, (short)0, O[rt][ct], false, false);
    }
  }

  // ---- epilogue: normalize by the row-sum tile + store ----
  float* ob = out + ((size_t)bh * NB + u) * BS * DD;
#pragma unroll
  for (int rt = 0; rt < 2; ++rt)
#pragma unroll
    for (int r = 0; r < 8; ++r) {
      const float inv = 1.0f / O[rt][4][r];   // replicated across all 16 lanes
      const int row = rows0 + rt * 16 + r + 8 * hf;
#pragma unroll
      for (int ct = 0; ct < 4; ++ct)
        ob[row * DD + ct * 16 + n] = O[rt][ct][r] * inv;
    }
}

extern "C" void kernel_launch(void* const* d_in, const int* in_sizes, int n_in,
                              void* d_out, int out_size, void* d_ws, size_t ws_size,
                              hipStream_t stream) {
  const float* q = (const float*)d_in[0];
  const float* k = (const float*)d_in[1];
  const float* v = (const float*)d_in[2];
  float* out = (float*)d_out;

  int*   ridx = (int*)d_ws;
  float* rval = (float*)((char*)d_ws + (size_t)BHN * NB * sizeof(int));

  hipFuncSetAttribute(reinterpret_cast<const void*>(sinkhorn_attn_kernel),
                      hipFuncAttributeMaxDynamicSharedMemorySize, SMEM_BYTES);

  sinkhorn_route_kernel<<<dim3(BHN), dim3(256), 0, stream>>>(q, k, ridx, rval);
  sinkhorn_attn_kernel<<<dim3(NB, BHN), dim3(256), SMEM_BYTES, stream>>>(
      q, k, v, ridx, rval, out);
}
